// GDPool_36739150250677
// MI455X (gfx1250) — compile-verified, matrix operand order checked by reading, not run on hardware
//
#include <hip/hip_runtime.h>

#define B_ 4
#define C_ 128
#define N_ 16384
#define S_ 2048
#define K_ 16
#define CH_ 4096
#define SWS 130   // padded LDS row stride for W (kills 16-way bank conflicts)

typedef __attribute__((ext_vector_type(2)))  float    v2f;
typedef __attribute__((ext_vector_type(8)))  float    v8f;
typedef __attribute__((ext_vector_type(16))) _Float16 v16h;

#define HAS_WMMA_F32X4 __has_builtin(__builtin_amdgcn_wmma_f32_16x16x4_f32)

// ---------------------------------------------------------------- init
__global__ __launch_bounds__(512) void k_init(float* vec) {
  vec[threadIdx.x] = 0.0f;   // B_*C_ = 512 entries; relu>=0 so int-bit atomicMax works
}

// ------------------------------------------------ fused GEMM + relu + col-max
// proj = relu(W @ Z + b); vector[b][o] = max_n proj[b][o][n]
// One block = 64 columns of one batch; 8 waves each own a 16x16 (o,n) WMMA tile x4.
__global__ __launch_bounds__(256) void k_proj_maxvec(
    const float* __restrict__ feats, const float* __restrict__ Wm,
    const float* __restrict__ bias, int* __restrict__ vecbits)
{
#if HAS_WMMA_F32X4
  __shared__ float    sW[C_ * SWS];  // ~66.6 KB  W[o][k]
  __shared__ float    sZ[C_ * 64];   //   32 KB   Z[k][n]
#else
  __shared__ _Float16 sW[C_ * SWS];
  __shared__ _Float16 sZ[C_ * 64];
#endif
  const int b   = blockIdx.y;
  const int n0  = blockIdx.x * 64;
  const int tid = threadIdx.x;

  for (int i = tid; i < C_ * C_; i += 256)
    sW[(i >> 7) * SWS + (i & 127)] = Wm[i];
  for (int i = tid; i < C_ * 64; i += 256) {
    int k = i >> 6, n = i & 63;
    sZ[i] = feats[(size_t)b * C_ * N_ + (size_t)k * N_ + n0 + n];
  }
  __syncthreads();

  const int wave = tid >> 5;
  const int lane = tid & 31;
  const int hl   = lane >> 4;     // which 16-lane half
  const int l16  = lane & 15;
  const int nsub = (wave & 3) * 16;

  for (int ob = 0; ob < 4; ++ob) {
    const int o0 = ((wave >> 2) * 4 + ob) * 16;
    v8f acc = {};
#if HAS_WMMA_F32X4
    // A 16x4 f32: lanes0-15 K={0,1}, lanes16-31 K={2,3}; B mirrors on rows.
    for (int k4 = 0; k4 < C_ / 4; ++k4) {
      const int ka = k4 * 4 + hl * 2;
      v2f a, bf;
      a.x  = sW[(o0 + l16) * SWS + ka];
      a.y  = sW[(o0 + l16) * SWS + ka + 1];
      bf.x = sZ[(size_t)(ka)     * 64 + nsub + l16];
      bf.y = sZ[(size_t)(ka + 1) * 64 + nsub + l16];
      acc = __builtin_amdgcn_wmma_f32_16x16x4_f32(false, a, false, bf,
                                                  (short)0, acc, false, false);
    }
#else
    // fallback: f16 16x16x32 WMMA (codegen-confirmed builtin)
    for (int kb = 0; kb < C_ / 32; ++kb) {
      v16h a, bf;
      #pragma unroll
      for (int j = 0; j < 16; ++j) {
        int kk = kb * 32 + ((j < 8) ? (hl * 8 + j) : (16 + hl * 8 + (j - 8)));
        a[j]  = sW[(o0 + l16) * SWS + kk];
        bf[j] = sZ[(size_t)kk * 64 + nsub + l16];
      }
      acc = __builtin_amdgcn_wmma_f32_16x16x32_f16(false, a, false, bf,
                                                   (short)0, acc, false, false);
    }
#endif
    // C/D layout: vgpr i, lane half hl -> (o = o0+i+hl*8, n = l16). Reduce over n.
    #pragma unroll
    for (int i = 0; i < 8; ++i) {
      const int o = o0 + i + hl * 8;
      float val = fmaxf(acc[i] + bias[o], 0.0f);
      #pragma unroll
      for (int off = 8; off > 0; off >>= 1)            // stays within 16-lane half
        val = fmaxf(val, __shfl_xor(val, off, 32));
      if (l16 == 0)
        atomicMax(&vecbits[b * C_ + o], __float_as_int(val));
    }
  }
}

// ---------------------------------------------------------------- scores
__global__ __launch_bounds__(256) void k_scores(
    const float* __restrict__ feats, const float* __restrict__ vec,
    float* __restrict__ scores)
{
  const int id = blockIdx.x * 256 + threadIdx.x;
  const int b  = id >> 14;
  const int n  = id & (N_ - 1);
  const float* fp = feats + (size_t)b * C_ * N_ + n;
  float w = 0.0f;
  #pragma unroll 8
  for (int c = 0; c < C_; ++c)
    w += fp[(size_t)c * N_] * vec[b * C_ + c];
  scores[id] = 1.0f / (1.0f + __expf(-w));
}

// --------------------------------------------- full in-LDS bitonic top-k sort
// 16384 u64 keys = 128 KB LDS; key = (~sortable(score) << 32) | idx
// -> ascending sort == score descending, ties by lower index (matches top_k).
__global__ __launch_bounds__(1024) void k_sort(
    const float* __restrict__ scores, int* __restrict__ idxs,
    float* __restrict__ vals)
{
  __shared__ unsigned long long keys[N_];
  const int b = blockIdx.x, tid = threadIdx.x;
  for (int i = tid; i < N_; i += 1024) {
    unsigned bits = __float_as_uint(scores[b * N_ + i]);
    unsigned u = (bits & 0x80000000u) ? ~bits : (bits | 0x80000000u);
    unsigned v = ~u;
    keys[i] = ((unsigned long long)v << 32) | (unsigned)i;
  }
  for (unsigned size = 2; size <= (unsigned)N_; size <<= 1) {
    for (unsigned stride = size >> 1; stride > 0; stride >>= 1) {
      __syncthreads();
      #pragma unroll
      for (int p = 0; p < N_ / 1024; ++p) {
        int i = tid + (p << 10);
        int partner = i ^ (int)stride;
        if (partner > i) {
          unsigned long long x = keys[i], y = keys[partner];
          bool asc = (((unsigned)i & size) == 0);
          if ((x > y) == asc) { keys[i] = y; keys[partner] = x; }
        }
      }
    }
  }
  __syncthreads();
  for (int i = tid; i < S_; i += 1024) {
    int idx = (int)(keys[i] & 0xFFFFFFFFull);
    idxs[b * S_ + i] = idx;
    vals[b * S_ + i] = scores[b * N_ + idx];
  }
}

// ------------------------------------- gather pool coords/feats (coalesced m)
__global__ __launch_bounds__(256) void k_gather1(
    const float* __restrict__ coords, const float* __restrict__ feats,
    const int* __restrict__ idxs, const float* __restrict__ vals,
    float* __restrict__ out0, float* __restrict__ out1,
    float* __restrict__ out2)
{
  const int c = blockIdx.x, b = blockIdx.y;
  for (int m = threadIdx.x; m < S_; m += 256) {
    const int   idx = idxs[b * S_ + m];
    const float v   = vals[b * S_ + m];
    out2[((size_t)b * 256 + c) * S_ + m] =
        feats[(size_t)b * C_ * N_ + (size_t)c * N_ + idx] * v;
    if (c < 3) {
      float pc = coords[(size_t)b * 3 * N_ + (size_t)c * N_ + idx];
      out0[((size_t)b * 3 + c) * S_ + m] = pc;
      out1[((size_t)b * 3 + c) * S_ + m] = pc * v;
    }
  }
}

// ----------------------------- kNN: per-chunk top-16 with LDS-staged points
__global__ __launch_bounds__(256) void k_knn_part(
    const float* __restrict__ coords, const float* __restrict__ nodes,
    float* __restrict__ pd, int* __restrict__ pi)
{
  __shared__ float sx[CH_], sy[CH_], sz[CH_];   // 48 KB
  const int b = blockIdx.z, chunk = blockIdx.y;
  const int m = blockIdx.x * 256 + threadIdx.x;
  const float nx = nodes[((size_t)b * 3 + 0) * S_ + m];
  const float ny = nodes[((size_t)b * 3 + 1) * S_ + m];
  const float nz = nodes[((size_t)b * 3 + 2) * S_ + m];
  const float sn = nx * nx + ny * ny + nz * nz;
  const size_t base = (size_t)b * 3 * N_;
  const int j0 = chunk * CH_;
  for (int i = threadIdx.x; i < CH_; i += 256) {
    sx[i] = coords[base + 0 * N_ + j0 + i];
    sy[i] = coords[base + 1 * N_ + j0 + i];
    sz[i] = coords[base + 2 * N_ + j0 + i];
  }
  __syncthreads();

  float bd[K_]; int bi[K_];
  #pragma unroll
  for (int t = 0; t < K_; ++t) { bd[t] = 3.4e38f; bi[t] = 0; }
  float cmax = 3.4e38f; int cslot = 0;

  for (int j = 0; j < CH_; ++j) {            // wave-uniform LDS reads: broadcast
    const float x = sx[j], y = sy[j], z = sz[j];
    const float d = (x * x + y * y + z * z) + sn
                    - 2.0f * (x * nx + y * ny + z * nz);
    if (d < cmax) {
      #pragma unroll
      for (int t = 0; t < K_; ++t)           // static-indexed insert, no spills
        if (t == cslot) { bd[t] = d; bi[t] = j0 + j; }
      cmax = bd[0]; cslot = 0;
      #pragma unroll
      for (int t = 1; t < K_; ++t)
        if (bd[t] > cmax) { cmax = bd[t]; cslot = t; }
    }
  }
  const size_t o = (((size_t)b * S_ + m) * 4 + chunk) * K_;
  #pragma unroll
  for (int t = 0; t < K_; ++t) { pd[o + t] = bd[t]; pi[o + t] = bi[t]; }
}

// ------------------------------------------- merge 4 partial top-16 -> top-16
__global__ __launch_bounds__(256) void k_knn_merge(
    const float* __restrict__ pd, const int* __restrict__ pi,
    int* __restrict__ nn)
{
  const int g = blockIdx.x * 256 + threadIdx.x;   // b*S + m
  float bd[K_]; int bi[K_];
  #pragma unroll
  for (int t = 0; t < K_; ++t) { bd[t] = 3.4e38f; bi[t] = 0; }
  float cmax = 3.4e38f; int cslot = 0;
  for (int q = 0; q < 4 * K_; ++q) {
    const float d  = pd[(size_t)g * 64 + q];
    const int   id = pi[(size_t)g * 64 + q];
    if (d < cmax) {
      #pragma unroll
      for (int t = 0; t < K_; ++t)
        if (t == cslot) { bd[t] = d; bi[t] = id; }
      cmax = bd[0]; cslot = 0;
      #pragma unroll
      for (int t = 1; t < K_; ++t)
        if (bd[t] > cmax) { cmax = bd[t]; cslot = t; }
    }
  }
  #pragma unroll
  for (int t = 0; t < K_; ++t) nn[(size_t)g * K_ + t] = bi[t];
}

// ------------------------------------------------- neighbor feature max-agg
__global__ __launch_bounds__(256) void k_agg(
    const float* __restrict__ feats, const int* __restrict__ nn,
    float* __restrict__ out2)
{
  const int c = blockIdx.x, b = blockIdx.y;
  for (int m = threadIdx.x; m < S_; m += 256) {
    const size_t g = (size_t)b * S_ + m;
    float acc = -3.4e38f;
    #pragma unroll
    for (int j = 0; j < K_; ++j) {
      const int idx = nn[g * K_ + j];
      acc = fmaxf(acc, feats[(size_t)b * C_ * N_ + (size_t)c * N_ + idx]);
    }
    out2[((size_t)b * 256 + 128 + c) * S_ + m] = acc;
  }
}

// ----------------------------------------------------------------- launch
extern "C" void kernel_launch(void* const* d_in, const int* in_sizes, int n_in,
                              void* d_out, int out_size, void* d_ws, size_t ws_size,
                              hipStream_t stream)
{
  (void)in_sizes; (void)n_in; (void)out_size; (void)ws_size;
  const float* coords = (const float*)d_in[0];
  const float* feats  = (const float*)d_in[1];
  const float* Wm     = (const float*)d_in[2];
  const float* bias   = (const float*)d_in[3];

  float* vec    = (float*)d_ws;                  // B*C floats (bits via atomicMax)
  float* scores = vec + B_ * C_;                 // B*N
  int*   idxs   = (int*)(scores + B_ * N_);      // B*S
  float* vals   = (float*)(idxs + B_ * S_);      // B*S
  float* pd     = vals + B_ * S_;                // B*S*64
  int*   pi     = (int*)(pd + (size_t)B_ * S_ * 64); // B*S*64
  int*   nn     = pi + (size_t)B_ * S_ * 64;     // B*S*K

  float* out0 = (float*)d_out;                   // pool_coords_static (B,3,S)
  float* out1 = out0 + (size_t)B_ * 3 * S_;      // pool_coords        (B,3,S)
  float* out2 = out1 + (size_t)B_ * 3 * S_;      // pool_feats         (B,256,S)

  k_init       <<<1, 512, 0, stream>>>(vec);
  k_proj_maxvec<<<dim3(N_ / 64, B_), 256, 0, stream>>>(feats, Wm, bias, (int*)vec);
  k_scores     <<<(B_ * N_) / 256, 256, 0, stream>>>(feats, vec, scores);
  k_sort       <<<B_, 1024, 0, stream>>>(scores, idxs, vals);
  k_gather1    <<<dim3(C_, B_), 256, 0, stream>>>(coords, feats, idxs, vals,
                                                  out0, out1, out2);
  k_knn_part   <<<dim3(S_ / 256, 4, B_), 256, 0, stream>>>(coords, out0, pd, pi);
  k_knn_merge  <<<(B_ * S_) / 256, 256, 0, stream>>>(pd, pi, nn);
  k_agg        <<<dim3(C_, B_), 256, 0, stream>>>(feats, nn, out2);
}
